// FasterRCNN_2190433321277
// MI455X (gfx1250) — matrix operand order repaired and checked by reading, not compiled, not run on hardware
//
#include <hip/hip_runtime.h>
#include <hip/hip_bf16.h>

// ---------------------------------------------------------------------------
// Faster R-CNN forward for MI455X (gfx1250, wave32, WMMA).
// Split-precision GEMM: fp32 operands -> bf16 hi+lo, 3 WMMAs per product
// (hh + hl + lh), fp32 accumulate.  HBM floor ~520MB weights @ 23.3TB/s;
// double-buffered LDS ping-pong overlaps the weight stream with the WMMA
// pipe, one barrier per 32-K step, 6 WMMAs per step per wave (TN=32).
// ---------------------------------------------------------------------------

typedef __attribute__((ext_vector_type(16))) __bf16 v16bf;
typedef __attribute__((ext_vector_type(8)))  __bf16 v8bf;
typedef __attribute__((ext_vector_type(8)))  float  v8f;

#define TM      128   // M rows per block (8 waves x 16)
#define TN      32    // N cols per block (2 accumulators per wave)
#define APITCH  40    // LDS row pitch in bf16 (80B: bank-conflict-free, 16B aligned)

static __device__ __forceinline__ void split_bf16(float v,
                                                  unsigned short& hi,
                                                  unsigned short& lo) {
  union { float f; unsigned u; } c; c.f = v;
  hi = (unsigned short)(c.u >> 16);
  union { unsigned u; float f; } h; h.u = c.u & 0xFFFF0000u;
  union { float f; unsigned u; } r; r.f = v - h.f;
  lo = (unsigned short)(r.u >> 16);
}

static __device__ __forceinline__ v16bf cat8(v8bf a, v8bf b) {
  v16bf r;
#pragma unroll
  for (int i = 0; i < 8; ++i) { r[i] = a[i]; r[i + 8] = b[i]; }
  return r;
}

// C[M,N] = A(MxK) * Bw(NxK)^T + bias, optional ReLU.
// amode==0: A[m*a_rs + k*a_cs];  amode==1: im2col gather from x[512,37,50].
// Output store: out[m*out_rs + n*out_cs].
__global__ __launch_bounds__(256) void gemm_bf16x2_wmma(
    const float* __restrict__ A, const float* __restrict__ Bw,
    const float* __restrict__ bias, float* __restrict__ out,
    int M, int N, int K, int a_rs, int a_cs,
    int out_rs, int out_cs, int do_relu, int amode)
{
  __shared__ unsigned short Ah[2][TM * APITCH];
  __shared__ unsigned short Al[2][TM * APITCH];
  __shared__ unsigned short Bh[2][TN * APITCH];
  __shared__ unsigned short Bl[2][TN * APITCH];

  const int tid   = threadIdx.x;
  const int lane  = tid & 31;
  const int wave  = tid >> 5;
  const int mbase = blockIdx.y * TM;
  const int nbase = blockIdx.x * TN;

  v8f acc0 = {0.f, 0.f, 0.f, 0.f, 0.f, 0.f, 0.f, 0.f};
  v8f acc1 = {0.f, 0.f, 0.f, 0.f, 0.f, 0.f, 0.f, 0.f};

  float areg[16];   // global->register staging (A: TM*32/256 per thread)
  float breg[4];    // (B: TN*32/256 per thread)

  // ---- global -> registers (overlaps with WMMA of previous tile) ----
  auto load_tile = [&](int kt) {
#pragma unroll
    for (int i = 0; i < 16; ++i) {
      int idx = tid + i * 256;
      int row = idx >> 5, col = idx & 31;
      int m = mbase + row, k = kt + col;
      float v = 0.f;
      if (amode == 0) {
        if (m < M) v = A[(size_t)m * a_rs + (size_t)k * a_cs];
      } else {
        if (m < M) {
          int ci = k / 9;  int r9 = k - ci * 9;
          int ky = r9 / 3; int kx = r9 - ky * 3;
          int py = m / 50; int px = m - py * 50;
          int iy = py + ky - 1, ix = px + kx - 1;
          if (iy >= 0 && iy < 37 && ix >= 0 && ix < 50)
            v = A[ci * 1850 + iy * 50 + ix];
        }
      }
      areg[i] = v;
    }
#pragma unroll
    for (int i = 0; i < 4; ++i) {
      int idx = tid + i * 256;
      int row = idx >> 5, col = idx & 31;
      int n = nbase + row, k = kt + col;
      float v = 0.f;
      if (n < N) {
        v = Bw[(size_t)n * K + k];
        if (kt + 1024 < K)   // stream-ahead on the HBM-bound weight path
          __builtin_prefetch(&Bw[(size_t)n * K + k + 1024], 0, 0);
      }
      breg[i] = v;
    }
  };

  // ---- registers -> LDS (bf16 hi/lo split), into buffer b ----
  auto store_tile = [&](int b) {
#pragma unroll
    for (int i = 0; i < 16; ++i) {
      int idx = tid + i * 256;
      int row = idx >> 5, col = idx & 31;
      unsigned short h, l; split_bf16(areg[i], h, l);
      Ah[b][row * APITCH + col] = h;
      Al[b][row * APITCH + col] = l;
    }
#pragma unroll
    for (int i = 0; i < 4; ++i) {
      int idx = tid + i * 256;
      int row = idx >> 5, col = idx & 31;
      unsigned short h, l; split_bf16(breg[i], h, l);
      Bh[b][row * APITCH + col] = h;
      Bl[b][row * APITCH + col] = l;
    }
  };

  // prologue: fill buffer 0
  load_tile(0);
  store_tile(0);
  __syncthreads();

  for (int kt = 0; kt < K; kt += 32) {
    const int  cur      = (kt >> 5) & 1;
    const bool has_next = (kt + 32) < K;
    if (has_next) load_tile(kt + 32);   // issue next global loads first

    // ---- fragments per CDNA5 16-bit layouts ----
    // A 16x32: lanes 0-15 row=lane, K {0..7}|{16..23}; lanes 16-31 K {8..15}|{24..31}
    const int l15  = lane & 15;
    const int arow = wave * 16 + l15;
    const int c0   = (lane & 16) ? 8 : 0;
    const unsigned short* AhC = Ah[cur];
    const unsigned short* AlC = Al[cur];
    const unsigned short* BhC = Bh[cur];
    const unsigned short* BlC = Bl[cur];
    v16bf aH = cat8(*(const v8bf*)(AhC + arow * APITCH + c0),
                    *(const v8bf*)(AhC + arow * APITCH + c0 + 16));
    v16bf aL = cat8(*(const v8bf*)(AlC + arow * APITCH + c0),
                    *(const v8bf*)(AlC + arow * APITCH + c0 + 16));
    // B 32x16: lanes 0-15 col=lane K {0..7}|{16..23}; lanes 16-31 col=lane-16 K {8..15}|{24..31}
    const int kb = (lane & 16) ? 8 : 0;
    v16bf bH0 = cat8(*(const v8bf*)(BhC + l15 * APITCH + kb),
                     *(const v8bf*)(BhC + l15 * APITCH + kb + 16));
    v16bf bL0 = cat8(*(const v8bf*)(BlC + l15 * APITCH + kb),
                     *(const v8bf*)(BlC + l15 * APITCH + kb + 16));
    v16bf bH1 = cat8(*(const v8bf*)(BhC + (16 + l15) * APITCH + kb),
                     *(const v8bf*)(BhC + (16 + l15) * APITCH + kb + 16));
    v16bf bL1 = cat8(*(const v8bf*)(BlC + (16 + l15) * APITCH + kb),
                     *(const v8bf*)(BlC + (16 + l15) * APITCH + kb + 16));

    // split-precision products: hh + hl + lh (lo*lo ~2^-18 rel, dropped)
    acc0 = __builtin_amdgcn_wmma_f32_16x16x32_bf16(false, aH, false, bH0, (short)0, acc0, false, false);
    acc0 = __builtin_amdgcn_wmma_f32_16x16x32_bf16(false, aH, false, bL0, (short)0, acc0, false, false);
    acc0 = __builtin_amdgcn_wmma_f32_16x16x32_bf16(false, aL, false, bH0, (short)0, acc0, false, false);
    acc1 = __builtin_amdgcn_wmma_f32_16x16x32_bf16(false, aH, false, bH1, (short)0, acc1, false, false);
    acc1 = __builtin_amdgcn_wmma_f32_16x16x32_bf16(false, aH, false, bL1, (short)0, acc1, false, false);
    acc1 = __builtin_amdgcn_wmma_f32_16x16x32_bf16(false, aL, false, bH1, (short)0, acc1, false, false);

    if (has_next) store_tile(cur ^ 1);  // ping-pong: write the other buffer
    __syncthreads();                    // one barrier per K-step
  }

  // ---- epilogue: C layout VGPR r: lanes0-15 M=r, lanes16-31 M=r+8; N=lane%16
  const int l15   = lane & 15;
  const int coln0 = nbase + l15;
  const int coln1 = coln0 + 16;
  const float bv0 = (coln0 < N) ? bias[coln0] : 0.f;
  const float bv1 = (coln1 < N) ? bias[coln1] : 0.f;
  const int rbase = mbase + wave * 16 + ((lane & 16) ? 8 : 0);
#pragma unroll
  for (int r = 0; r < 8; ++r) {
    int row = rbase + r;
    float v0 = acc0[r] + bv0;
    float v1 = acc1[r] + bv1;
    if (do_relu) { v0 = fmaxf(v0, 0.f); v1 = fmaxf(v1, 0.f); }
    if (row < M && coln0 < N)
      out[(size_t)row * out_rs + (size_t)coln0 * out_cs] = v0;
    if (row < M && coln1 < N)
      out[(size_t)row * out_rs + (size_t)coln1 * out_cs] = v1;
  }
}

// RoI adaptive max-pool, exact reference arithmetic (trunc scale, inclusive
// crop, ceil/floor bin edges, -FLT_MAX for clamped-empty bins).
__global__ __launch_bounds__(256) void roi_pool_kernel(
    const float* __restrict__ x, const float* __restrict__ rois,
    float* __restrict__ pooled)
{
  const int idx = blockIdx.x * 256 + threadIdx.x;
  const int total = 300 * 512 * 49;
  if (idx >= total) return;
  const int px = idx % 7;
  const int py = (idx / 7) % 7;
  const int c  = (idx / 49) % 512;
  const int n  = idx / (49 * 512);
  const float wscale = (float)(50.0 / 800.0);
  const float hscale = (float)(37.0 / 600.0);
  const int rx = (int)(rois[n * 4 + 0] * wscale);
  const int ry = (int)(rois[n * 4 + 1] * hscale);
  const int rw = (int)(rois[n * 4 + 2] * wscale);
  const int rh = (int)(rois[n * 4 + 3] * hscale);
  const int sh = rh + 1, sw = rw + 1;
  int ys = ry + (py * sh) / 7;
  int ye = ry + ((py + 1) * sh + 6) / 7; if (ye > 37) ye = 37;
  int xs = rx + (px * sw) / 7;
  int xe = rx + ((px + 1) * sw + 6) / 7; if (xe > 50) xe = 50;
  float m = -3.402823466e38f;
  for (int y = ys; y < ye; ++y)
    for (int xx = xs; xx < xe; ++xx)
      m = fmaxf(m, x[c * 1850 + y * 50 + xx]);
  pooled[idx] = m;   // layout [n][c][py][px] == fc1 K-order c*49+py*7+px
}

extern "C" void kernel_launch(void* const* d_in, const int* in_sizes, int n_in,
                              void* d_out, int out_size, void* d_ws, size_t ws_size,
                              hipStream_t stream)
{
  (void)in_sizes; (void)n_in; (void)out_size; (void)ws_size;
  const float* x    = (const float*)d_in[0];
  // d_in[1] = img: only its (fixed) shape is used -> scales hardcoded
  const float* rois = (const float*)d_in[2];
  const float* cfw  = (const float*)d_in[3];
  const float* cfb  = (const float*)d_in[4];
  const float* ccw  = (const float*)d_in[5];
  const float* ccb  = (const float*)d_in[6];
  const float* crw  = (const float*)d_in[7];
  const float* crb  = (const float*)d_in[8];
  const float* f1w  = (const float*)d_in[9];
  const float* f1b  = (const float*)d_in[10];
  const float* f2w  = (const float*)d_in[11];
  const float* f2b  = (const float*)d_in[12];
  const float* clw  = (const float*)d_in[13];
  const float* clb  = (const float*)d_in[14];
  const float* rgw  = (const float*)d_in[15];
  const float* rgb  = (const float*)d_in[16];
  float* out = (float*)d_out;

  // workspace layout (~44 MB, L2-resident activations)
  float* f      = (float*)d_ws;            // [512][1850]  conv feature (transposed C)
  float* pooled = f + 512 * 1850;          // [300][25088]
  float* h1     = pooled + 300 * 25088;    // [300][4096]
  float* h2     = h1 + 300 * 4096;         // [300][4096]

  dim3 blk(256);

  // 3x3 conv 512->512 + ReLU, implicit im2col GEMM: M=1850,N=512,K=4608;
  // store transposed so f is [C][H*W].
  gemm_bf16x2_wmma<<<dim3(16, 15), blk, 0, stream>>>(
      x, cfw, cfb, f, 1850, 512, 4608, 0, 0, /*out*/ 1, 1850, 1, /*im2col*/ 1);

  // RPN 1x1 convs on f^T (A strided: a_rs=1, a_cs=1850), outputs [A][HW]
  gemm_bf16x2_wmma<<<dim3(1, 15), blk, 0, stream>>>(
      f, ccw, ccb, out,          1850, 18, 512, 1, 1850, 1, 1850, 0, 0);
  gemm_bf16x2_wmma<<<dim3(2, 15), blk, 0, stream>>>(
      f, crw, crb, out + 33300,  1850, 36, 512, 1, 1850, 1, 1850, 0, 0);

  // RoI max pool -> pooled [300,512,7,7]
  roi_pool_kernel<<<dim3((300 * 512 * 49 + 255) / 256), blk, 0, stream>>>(
      x, rois, pooled);

  // VGG head: fc1 (M=300,N=4096,K=25088) + ReLU; fc2; cls/reg outputs
  gemm_bf16x2_wmma<<<dim3(128, 3), blk, 0, stream>>>(
      pooled, f1w, f1b, h1, 300, 4096, 25088, 25088, 1, 4096, 1, 1, 0);
  gemm_bf16x2_wmma<<<dim3(128, 3), blk, 0, stream>>>(
      h1, f2w, f2b, h2, 300, 4096, 4096, 4096, 1, 4096, 1, 1, 0);
  gemm_bf16x2_wmma<<<dim3(1, 3), blk, 0, stream>>>(
      h2, clw, clb, out + 99900,  300, 21, 4096, 4096, 1, 21, 1, 0, 0);
  gemm_bf16x2_wmma<<<dim3(3, 3), blk, 0, stream>>>(
      h2, rgw, rgb, out + 106200, 300, 84, 4096, 4096, 1, 84, 1, 0, 0);
}
